// TensorNet_42125039239893
// MI455X (gfx1250) — matrix operand order repaired and checked by reading, not compiled, not run on hardware
//
#include <hip/hip_runtime.h>

// ---------------- problem constants ----------------
#define E_EDGES 65536
#define NN 8192
#define NGRAPH 64
#define H 64
#define RB 32

typedef _Float16 half_t;
typedef __attribute__((ext_vector_type(16))) _Float16 v16h;
typedef __attribute__((ext_vector_type(8)))  _Float16 v8h;
typedef __attribute__((ext_vector_type(8)))  float    v8f;

// =====================================================================
// Generic WMMA GEMM:  out[m,n] = act( sum_k A[m,k]*W[n,k] + bias[n] ) * rowScale[m]
// A: [M,K] f16 row-major.  W: [N,K] f16 row-major ("out x in", lin()'s layout).
// One wave owns a 16-row strip of A and sweeps ALL N/16 output tiles (NT),
// holding the A fragment in registers -> A is loaded from memory exactly once.
// 4 waves per block along M.  M%64==0, N==NT*16, K%32==0.
// =====================================================================
template <int NT>
__global__ __launch_bounds__(128) void k_gemm(
    const half_t* __restrict__ A, const half_t* __restrict__ W,
    const float* __restrict__ bias, const float* __restrict__ rowScale,
    float* __restrict__ outF, half_t* __restrict__ outH,
    int M, int K, int act)
{
    const int lane  = threadIdx.x & 31;
    const int wave  = threadIdx.x >> 5;
    const int tileM = blockIdx.x * 4 + wave;
    const int g = lane >> 4;      // lane group (0/1)
    const int l = lane & 15;
    const int N = NT * 16;

    const int aRowIdx = tileM * 16 + l;          // A row held by this lane's fragment
    const half_t* aRow = A + (size_t)aRowIdx * K;
    const half_t* wCol = W + (size_t)l * K;      // W row (= output column) base for B frags

    v8f acc[NT];
#pragma unroll
    for (int t = 0; t < NT; ++t) acc[t] = (v8f){};

    for (int k0 = 0; k0 < K; k0 += 32) {
        // A 16x32 f16 layout: lanes 0-15 row M=l; halves e -> K = g*16 + e (contiguous)
        v8h alo = *(const v8h*)(aRow + k0 + g * 16);
        v8h ahi = *(const v8h*)(aRow + k0 + g * 16 + 8);
        __builtin_prefetch(aRow + k0 + 64, 0, 0);   // global_prefetch_b8 on the A stream
        v16h a;
#pragma unroll
        for (int i = 0; i < 8; ++i) { a[i] = alo[i]; a[i + 8] = ahi[i]; }
        // B 32x16 f16 layout: lane n -> col n%16, halves e -> K = e + (n/16)*16 (contiguous)
#pragma unroll
        for (int t = 0; t < NT; ++t) {
            v16h b = *(const v16h*)(wCol + (size_t)t * 16 * K + k0 + g * 16);
            acc[t] = __builtin_amdgcn_wmma_f32_16x16x32_f16(
                false, a, false, b, (short)0, acc[t], false, false);
        }
    }

#pragma unroll
    for (int t = 0; t < NT; ++t) {
        const int n  = t * 16 + l;
        const float bv = bias ? bias[n] : 0.0f;
#pragma unroll
        for (int r = 0; r < 8; ++r) {
            int m = tileM * 16 + g * 8 + r;   // C/D layout: VGPR r -> M = r + g*8, N = l
            float v = acc[t][r] + bv;
            if (act == 1) v = v / (1.0f + __expf(-v));   // SiLU
            if (rowScale) v *= rowScale[m];
            size_t o = (size_t)m * N + n;
            if (outF) outF[o] = v;
            if (outH) outH[o] = (half_t)v;
        }
    }
}

// ---------------- small helper kernels ----------------
__global__ void k_f32_to_f16(const float* __restrict__ s, half_t* __restrict__ d, int n)
{
    int i = blockIdx.x * blockDim.x + threadIdx.x;
    if (i < n) d[i] = (half_t)s[i];
}

// edge geometry: cutoff C, unit vector vn, RBF embedding (f16)
__global__ void k_edge_geom(const float* __restrict__ pos, const float* __restrict__ cells,
                            const int* __restrict__ ei, const float* __restrict__ eo,
                            const int* __restrict__ batch,
                            float* __restrict__ Cc, float* __restrict__ vn,
                            half_t* __restrict__ eattr16)
{
    int e = blockIdx.x * blockDim.x + threadIdx.x;
    if (e >= E_EDGES) return;
    int s = ei[e];
    int d = ei[E_EDGES + e];
    const float* cell = cells + (size_t)batch[s] * 9;
    float o0 = eo[e * 3 + 0], o1 = eo[e * 3 + 1], o2 = eo[e * 3 + 2];
    float r0 = pos[d * 3 + 0] - pos[s * 3 + 0] + o0 * cell[0] + o1 * cell[3] + o2 * cell[6];
    float r1 = pos[d * 3 + 1] - pos[s * 3 + 1] + o0 * cell[1] + o1 * cell[4] + o2 * cell[7];
    float r2 = pos[d * 3 + 2] - pos[s * 3 + 2] + o0 * cell[2] + o1 * cell[5] + o2 * cell[8];
    float dd = sqrtf(r0 * r0 + r1 * r1 + r2 * r2);
    float dinv = dd > 0.0f ? 1.0f / dd : 0.0f;
    float cut = dd < 5.0f ? 0.5f * (cosf(3.14159265358979f * dd * 0.2f) + 1.0f) : 0.0f;
    Cc[e] = cut;
    vn[e * 3 + 0] = r0 * dinv; vn[e * 3 + 1] = r1 * dinv; vn[e * 3 + 2] = r2 * dinv;
    float start = __expf(-5.0f);
    float step  = (1.0f - start) / (RB - 1);
    float bet   = (2.0f / RB) * (1.0f - start);
    bet = 1.0f / (bet * bet);
    float ed = __expf(-dd);     // alpha = 5/CUTOFF = 1
    for (int r = 0; r < RB; ++r) {
        float t = ed - (start + r * step);
        eattr16[(size_t)e * RB + r] = (half_t)(cut * __expf(-bet * t * t));
    }
}

// Z[src]||Z[dst] gather -> f16 [E,128]
__global__ void k_gather_z(const int* __restrict__ ei, const int* __restrict__ numbers,
                           const float* __restrict__ embZ, half_t* __restrict__ zcat)
{
    int idx = blockIdx.x * blockDim.x + threadIdx.x;
    if (idx >= E_EDGES * 128) return;
    int e = idx >> 7, i = idx & 127;
    int node = (i < 64) ? ei[e] : ei[E_EDGES + e];
    zcat[idx] = (half_t)embZ[numbers[node] * H + (i & 63)];
}

// scatter In/An/Sn (compact: I=[N,1,H], A=[N,3,H], S=[N,6,H]) keyed by src
__global__ void k_embed_scatter(const int* __restrict__ ei, const float* __restrict__ Zij,
                                const float* __restrict__ W1, const float* __restrict__ W2,
                                const float* __restrict__ W3, const float* __restrict__ vn,
                                float* __restrict__ cI, float* __restrict__ cA, float* __restrict__ cS)
{
    int idx = blockIdx.x * blockDim.x + threadIdx.x;
    if (idx >= E_EDGES * H) return;
    int e = idx >> 6, h = idx & 63;
    int s = ei[e];
    float z  = Zij[idx];
    float m1 = z * W1[idx], m2 = z * W2[idx], m3 = z * W3[idx];
    float v0 = vn[e * 3 + 0], v1 = vn[e * 3 + 1], v2 = vn[e * 3 + 2];
    atomicAdd(cI + (size_t)s * H + h, m1);
    // skew(vn): a01=-v2, a02=v1, a12=-v0
    atomicAdd(cA + ((size_t)s * 3 + 0) * H + h, m2 * (-v2));
    atomicAdd(cA + ((size_t)s * 3 + 1) * H + h, m2 * ( v1));
    atomicAdd(cA + ((size_t)s * 3 + 2) * H + h, m2 * (-v0));
    // traceless sym outer product (|vn|=1)
    const float third = 1.0f / 3.0f;
    atomicAdd(cS + ((size_t)s * 6 + 0) * H + h, m3 * (v0 * v0 - third));
    atomicAdd(cS + ((size_t)s * 6 + 1) * H + h, m3 * (v0 * v1));
    atomicAdd(cS + ((size_t)s * 6 + 2) * H + h, m3 * (v0 * v2));
    atomicAdd(cS + ((size_t)s * 6 + 3) * H + h, m3 * (v1 * v1 - third));
    atomicAdd(cS + ((size_t)s * 6 + 4) * H + h, m3 * (v1 * v2));
    atomicAdd(cS + ((size_t)s * 6 + 5) * H + h, m3 * (v2 * v2 - third));
}

__device__ __forceinline__ void expand9(float i, const float a[3], const float s[6], float X[9])
{
    X[0] =  i + s[0]; X[1] =  a[0] + s[1]; X[2] =  a[1] + s[2];
    X[3] = -a[0] + s[1]; X[4] =  i + s[3]; X[5] =  a[2] + s[4];
    X[6] = -a[1] + s[2]; X[7] = -a[2] + s[4]; X[8] =  i + s[5];
}

// tensor_norm(In+An+Sn) + f16 copies of compacts for chan_lin GEMMs
__global__ void k_node_norm_embed(const float* __restrict__ cI, const float* __restrict__ cA,
                                  const float* __restrict__ cS, float* __restrict__ nrm,
                                  half_t* __restrict__ cI16, half_t* __restrict__ cA16,
                                  half_t* __restrict__ cS16)
{
    int idx = blockIdx.x * blockDim.x + threadIdx.x;
    if (idx >= NN * H) return;
    int n = idx >> 6, h = idx & 63;
    float i = cI[idx];
    float a[3], s[6], X[9];
#pragma unroll
    for (int c = 0; c < 3; ++c) a[c] = cA[((size_t)n * 3 + c) * H + h];
#pragma unroll
    for (int c = 0; c < 6; ++c) s[c] = cS[((size_t)n * 6 + c) * H + h];
    expand9(i, a, s, X);
    float t = 0.0f;
#pragma unroll
    for (int c = 0; c < 9; ++c) t += X[c] * X[c];
    nrm[idx] = t;
    cI16[idx] = (half_t)i;
#pragma unroll
    for (int c = 0; c < 3; ++c) cA16[((size_t)n * 3 + c) * H + h] = (half_t)a[c];
#pragma unroll
    for (int c = 0; c < 6; ++c) cS16[((size_t)n * 6 + c) * H + h] = (half_t)s[c];
}

// row-wise layernorm, f16 output (row lengths 64 / 192)
__global__ void k_layernorm(const float* __restrict__ x, const float* __restrict__ gg,
                            const float* __restrict__ bb, half_t* __restrict__ out,
                            int rows, int L)
{
    int r = blockIdx.x * blockDim.x + threadIdx.x;
    if (r >= rows) return;
    const float* p = x + (size_t)r * L;
    float mu = 0.0f;
    for (int i = 0; i < L; ++i) mu += p[i];
    mu /= L;
    float v = 0.0f;
    for (int i = 0; i < L; ++i) { float t = p[i] - mu; v += t * t; }
    v /= L;
    float inv = rsqrtf(v + 1e-5f);
    half_t* o = out + (size_t)r * L;
    for (int i = 0; i < L; ++i) o[i] = (half_t)((p[i] - mu) * inv * gg[i] + bb[i]);
}

// X = chanlin(I)*n0 + chanlin(A)*n1 + chanlin(S)*n2, stored [N,9,H]
__global__ void k_build_X(const float* __restrict__ dI, const float* __restrict__ dA,
                          const float* __restrict__ dS, const float* __restrict__ n3,
                          float* __restrict__ X)
{
    int idx = blockIdx.x * blockDim.x + threadIdx.x;
    if (idx >= NN * H) return;
    int n = idx >> 6, h = idx & 63;
    float w0 = n3[(size_t)n * 192 + h * 3 + 0];
    float w1 = n3[(size_t)n * 192 + h * 3 + 1];
    float w2 = n3[(size_t)n * 192 + h * 3 + 2];
    float i = dI[idx] * w0;
    float a[3], s[6], Xv[9];
#pragma unroll
    for (int c = 0; c < 3; ++c) a[c] = dA[((size_t)n * 3 + c) * H + h] * w1;
#pragma unroll
    for (int c = 0; c < 6; ++c) s[c] = dS[((size_t)n * 6 + c) * H + h] * w2;
    expand9(i, a, s, Xv);
#pragma unroll
    for (int c = 0; c < 9; ++c) X[((size_t)n * 9 + c) * H + h] = Xv[c];
}

// X <- X/(|X|^2+1) in place; emit compact f16 I/A/S of normalized X
__global__ void k_norm_decomp(float* __restrict__ X, half_t* __restrict__ cI16,
                              half_t* __restrict__ cA16, half_t* __restrict__ cS16)
{
    int idx = blockIdx.x * blockDim.x + threadIdx.x;
    if (idx >= NN * H) return;
    int n = idx >> 6, h = idx & 63;
    float x[9];
#pragma unroll
    for (int c = 0; c < 9; ++c) x[c] = X[((size_t)n * 9 + c) * H + h];
    float t = 0.0f;
#pragma unroll
    for (int c = 0; c < 9; ++c) t += x[c] * x[c];
    float sc = 1.0f / (t + 1.0f);
#pragma unroll
    for (int c = 0; c < 9; ++c) { x[c] *= sc; X[((size_t)n * 9 + c) * H + h] = x[c]; }
    float i = (x[0] + x[4] + x[8]) * (1.0f / 3.0f);
    cI16[idx] = (half_t)i;
    cA16[((size_t)n * 3 + 0) * H + h] = (half_t)(0.5f * (x[1] - x[3]));
    cA16[((size_t)n * 3 + 1) * H + h] = (half_t)(0.5f * (x[2] - x[6]));
    cA16[((size_t)n * 3 + 2) * H + h] = (half_t)(0.5f * (x[5] - x[7]));
    cS16[((size_t)n * 6 + 0) * H + h] = (half_t)(x[0] - i);
    cS16[((size_t)n * 6 + 1) * H + h] = (half_t)(0.5f * (x[1] + x[3]));
    cS16[((size_t)n * 6 + 2) * H + h] = (half_t)(0.5f * (x[2] + x[6]));
    cS16[((size_t)n * 6 + 3) * H + h] = (half_t)(x[4] - i);
    cS16[((size_t)n * 6 + 4) * H + h] = (half_t)(0.5f * (x[5] + x[7]));
    cS16[((size_t)n * 6 + 5) * H + h] = (half_t)(x[8] - i);
}

// Y = I + A + S (from chan_lin'd compacts), stored [N,9,H]
__global__ void k_build_Y(const float* __restrict__ dI, const float* __restrict__ dA,
                          const float* __restrict__ dS, float* __restrict__ Y)
{
    int idx = blockIdx.x * blockDim.x + threadIdx.x;
    if (idx >= NN * H) return;
    int n = idx >> 6, h = idx & 63;
    float a[3], s[6], Yv[9];
    float i = dI[idx];
#pragma unroll
    for (int c = 0; c < 3; ++c) a[c] = dA[((size_t)n * 3 + c) * H + h];
#pragma unroll
    for (int c = 0; c < 6; ++c) s[c] = dS[((size_t)n * 6 + c) * H + h];
    expand9(i, a, s, Yv);
#pragma unroll
    for (int c = 0; c < 9; ++c) Y[((size_t)n * 9 + c) * H + h] = Yv[c];
}

// msg[src] += ea0*I[dst] + ea1*A[dst] + ea2*S[dst]  (full 3x3)
__global__ void k_msg_scatter(const int* __restrict__ ei, const float* __restrict__ eaF,
                              const float* __restrict__ dI, const float* __restrict__ dA,
                              const float* __restrict__ dS, float* __restrict__ msg)
{
    int idx = blockIdx.x * blockDim.x + threadIdx.x;
    if (idx >= E_EDGES * H) return;
    int e = idx >> 6, h = idx & 63;
    int s = ei[e];
    int d = ei[E_EDGES + e];
    float e0 = eaF[(size_t)e * 192 + h * 3 + 0];
    float e1 = eaF[(size_t)e * 192 + h * 3 + 1];
    float e2 = eaF[(size_t)e * 192 + h * 3 + 2];
    float i = dI[(size_t)d * H + h];
    float a[3], sx[6], V[9];
#pragma unroll
    for (int c = 0; c < 3; ++c) a[c] = dA[((size_t)d * 3 + c) * H + h] * e1;
#pragma unroll
    for (int c = 0; c < 6; ++c) sx[c] = dS[((size_t)d * 6 + c) * H + h] * e2;
    expand9(i * e0, a, sx, V);
#pragma unroll
    for (int c = 0; c < 9; ++c) atomicAdd(msg + ((size_t)s * 9 + c) * H + h, V[c]);
}

// B = msg@Y + Y@msg ; compacts of B / (|B|^2 + 1) -> f16
__global__ void k_conv_update1(const float* __restrict__ msg, const float* __restrict__ Y,
                               half_t* __restrict__ cI16, half_t* __restrict__ cA16,
                               half_t* __restrict__ cS16)
{
    int idx = blockIdx.x * blockDim.x + threadIdx.x;
    if (idx >= NN * H) return;
    int n = idx >> 6, h = idx & 63;
    float M[9], Yv[9], B[9];
#pragma unroll
    for (int c = 0; c < 9; ++c) { M[c] = msg[((size_t)n * 9 + c) * H + h];
                                  Yv[c] = Y[((size_t)n * 9 + c) * H + h]; }
#pragma unroll
    for (int r = 0; r < 3; ++r)
#pragma unroll
        for (int c = 0; c < 3; ++c) {
            float acc = 0.0f;
#pragma unroll
            for (int k = 0; k < 3; ++k)
                acc += M[r * 3 + k] * Yv[k * 3 + c] + Yv[r * 3 + k] * M[k * 3 + c];
            B[r * 3 + c] = acc;
        }
    float t = 0.0f;
#pragma unroll
    for (int c = 0; c < 9; ++c) t += B[c] * B[c];
    float inv = 1.0f / (t + 1.0f);
#pragma unroll
    for (int c = 0; c < 9; ++c) B[c] *= inv;
    float i = (B[0] + B[4] + B[8]) * (1.0f / 3.0f);
    cI16[idx] = (half_t)i;
    cA16[((size_t)n * 3 + 0) * H + h] = (half_t)(0.5f * (B[1] - B[3]));
    cA16[((size_t)n * 3 + 1) * H + h] = (half_t)(0.5f * (B[2] - B[6]));
    cA16[((size_t)n * 3 + 2) * H + h] = (half_t)(0.5f * (B[5] - B[7]));
    cS16[((size_t)n * 6 + 0) * H + h] = (half_t)(B[0] - i);
    cS16[((size_t)n * 6 + 1) * H + h] = (half_t)(0.5f * (B[1] + B[3]));
    cS16[((size_t)n * 6 + 2) * H + h] = (half_t)(0.5f * (B[2] + B[6]));
    cS16[((size_t)n * 6 + 3) * H + h] = (half_t)(B[4] - i);
    cS16[((size_t)n * 6 + 4) * H + h] = (half_t)(0.5f * (B[5] + B[7]));
    cS16[((size_t)n * 6 + 5) * H + h] = (half_t)(B[8] - i);
}

// X += dX + dX@dX
__global__ void k_conv_update2(const float* __restrict__ dI, const float* __restrict__ dA,
                               const float* __restrict__ dS, float* __restrict__ X)
{
    int idx = blockIdx.x * blockDim.x + threadIdx.x;
    if (idx >= NN * H) return;
    int n = idx >> 6, h = idx & 63;
    float a[3], s[6], dX[9];
    float i = dI[idx];
#pragma unroll
    for (int c = 0; c < 3; ++c) a[c] = dA[((size_t)n * 3 + c) * H + h];
#pragma unroll
    for (int c = 0; c < 6; ++c) s[c] = dS[((size_t)n * 6 + c) * H + h];
    expand9(i, a, s, dX);
#pragma unroll
    for (int r = 0; r < 3; ++r)
#pragma unroll
        for (int c = 0; c < 3; ++c) {
            float acc = 0.0f;
#pragma unroll
            for (int k = 0; k < 3; ++k) acc += dX[r * 3 + k] * dX[k * 3 + c];
            size_t o = ((size_t)n * 9 + r * 3 + c) * H + h;
            X[o] = X[o] + dX[r * 3 + c] + acc;
        }
}

// feat = [|I|^2, |A|^2, |S|^2] per channel -> [N, 192]
__global__ void k_readout_feat(const float* __restrict__ X, float* __restrict__ feat)
{
    int idx = blockIdx.x * blockDim.x + threadIdx.x;
    if (idx >= NN * H) return;
    int n = idx >> 6, h = idx & 63;
    float x[9];
#pragma unroll
    for (int c = 0; c < 9; ++c) x[c] = X[((size_t)n * 9 + c) * H + h];
    float i  = (x[0] + x[4] + x[8]) * (1.0f / 3.0f);
    float a0 = 0.5f * (x[1] - x[3]), a1 = 0.5f * (x[2] - x[6]), a2 = 0.5f * (x[5] - x[7]);
    float s0 = x[0] - i, s1 = 0.5f * (x[1] + x[3]), s2 = 0.5f * (x[2] + x[6]);
    float s3 = x[4] - i, s4 = 0.5f * (x[5] + x[7]), s5 = x[8] - i;
    feat[(size_t)n * 192 + h]        = 3.0f * i * i;
    feat[(size_t)n * 192 + 64 + h]   = 2.0f * (a0 * a0 + a1 * a1 + a2 * a2);
    feat[(size_t)n * 192 + 128 + h]  = s0 * s0 + s3 * s3 + s5 * s5
                                     + 2.0f * (s1 * s1 + s2 * s2 + s4 * s4);
}

// final per-node energy + segment sum over graphs
__global__ void k_final(const float* __restrict__ h2, const float* __restrict__ w3,
                        const float* __restrict__ b3, const int* __restrict__ batch,
                        float* __restrict__ out)
{
    int n = blockIdx.x * blockDim.x + threadIdx.x;
    if (n >= NN) return;
    float s = b3[0];
    for (int h = 0; h < H; ++h) s += h2[(size_t)n * H + h] * w3[h];
    atomicAdd(out + batch[n], s);
}

// =====================================================================
extern "C" void kernel_launch(void* const* d_in, const int* in_sizes, int n_in,
                              void* d_out, int out_size, void* d_ws, size_t ws_size,
                              hipStream_t stream)
{
    (void)in_sizes; (void)n_in; (void)out_size;
    const float* positions = (const float*)d_in[0];
    const float* cells     = (const float*)d_in[1];
    const int*   numbers   = (const int*)d_in[2];
    const int*   ei        = (const int*)d_in[3];
    const float* eo        = (const float*)d_in[4];
    const int*   batch     = (const int*)d_in[5];
#define P(i) ((const float*)d_in[(i)])
    // param index map (insertion-order flatten of params dict):
    // 6 emb_Z | 7 emb2_W 8 emb2_b | 9/10 dp1 | 11/12 dp2 | 13/14 dp3 | 15/16 init_norm g/b
    // 17/18 emb_s1 | 19/20 emb_s2 | 21 emb_t | 22..28 conv0 | 29..35 conv1
    // 36/37 ln g/b | 38/39 mlp1 | 40/41 mlp2 | 42/43 mlp3

    // ---------------- workspace layout ----------------
    char* ws = (char*)d_ws;
    size_t off = 0;
    auto take = [&](size_t bytes) -> char* {
        char* p = ws + off; off = (off + bytes + 255) & ~(size_t)255; return p;
    };
    float*  bC      = (float*) take((size_t)E_EDGES * 4);
    float*  bVn     = (float*) take((size_t)E_EDGES * 12);
    half_t* bEattr  = (half_t*)take((size_t)E_EDGES * RB * 2);
    half_t* bZcat   = (half_t*)take((size_t)E_EDGES * 128 * 2);   // alias: ea1 f16 [E,64]
    float*  bZij    = (float*) take((size_t)E_EDGES * 64 * 4);    // alias: ea2 f16 [E,128]
    float*  bW1     = (float*) take((size_t)E_EDGES * 64 * 4 * 3);// alias: eaF f32 [E,192]
    float*  bW2 = bW1 + (size_t)E_EDGES * 64;
    float*  bW3 = bW2 + (size_t)E_EDGES * 64;
    float*  bcI     = (float*) take((size_t)NN * 64 * 4);
    float*  bcA     = (float*) take((size_t)NN * 192 * 4);
    float*  bcS     = (float*) take((size_t)NN * 384 * 4);
    half_t* bcI16   = (half_t*)take((size_t)NN * 64 * 2);
    half_t* bcA16   = (half_t*)take((size_t)NN * 192 * 2);
    half_t* bcS16   = (half_t*)take((size_t)NN * 384 * 2);
    float*  bdI     = (float*) take((size_t)NN * 64 * 4);
    float*  bdA     = (float*) take((size_t)NN * 192 * 4);
    float*  bdS     = (float*) take((size_t)NN * 384 * 4);
    float*  bNorm   = (float*) take((size_t)NN * 64 * 4);         // alias: h2 f32 [N,64]
    half_t* bLn16   = (half_t*)take((size_t)NN * 64 * 2);         // alias: h1 f16 [N,64]
    half_t* bM1_16  = (half_t*)take((size_t)NN * 128 * 2);
    float*  bM2f    = (float*) take((size_t)NN * 192 * 4);        // alias: feat f32 [N,192]
    float*  bX      = (float*) take((size_t)NN * 576 * 4);
    float*  bY      = (float*) take((size_t)NN * 576 * 4);
    float*  bMsg    = (float*) take((size_t)NN * 576 * 4);
    half_t* bFeat16 = (half_t*)take((size_t)NN * 192 * 2);
    half_t* whBase  = (half_t*)take((size_t)194560 * 2);
    if (off > ws_size) return;   // workspace too small: bail deterministically

    half_t* bEa1  = bZcat;           // [E,64] f16
    half_t* bEa2  = (half_t*)bZij;   // [E,128] f16
    float*  bEaF  = bW1;             // [E,192] f32
    float*  bFeat = bM2f;            // [N,192] f32
    half_t* bH1   = bLn16;           // [N,64] f16
    float*  bH2   = bNorm;           // [N,64] f32

    // ---------------- f16 weight staging ----------------
    size_t woff = 0;
    auto cvt = [&](int idx, int n) -> half_t* {
        half_t* dst = whBase + woff; woff += (size_t)n;
        k_f32_to_f16<<<(n + 255) / 256, 256, 0, stream>>>((const float*)d_in[idx], dst, n);
        return dst;
    };
    half_t* w_emb2 = cvt(7, 8192);
    half_t* w_dp1  = cvt(9, 2048);
    half_t* w_dp2  = cvt(11, 2048);
    half_t* w_dp3  = cvt(13, 2048);
    half_t* w_s1   = cvt(17, 8192);
    half_t* w_s2   = cvt(19, 24576);
    half_t* w_embt = cvt(21, 12288);
    half_t *w_cs1[2], *w_cs2[2], *w_cs3[2], *w_ct[2];
    for (int l = 0; l < 2; ++l) {
        int b = 22 + l * 7;
        w_cs1[l] = cvt(b + 0, 2048);
        w_cs2[l] = cvt(b + 2, 8192);
        w_cs3[l] = cvt(b + 4, 24576);
        w_ct[l]  = cvt(b + 6, 24576);
    }
    half_t* w_mlp1 = cvt(38, 12288);
    half_t* w_mlp2 = cvt(40, 4096);

    // GEMM launcher: NT = N/16 selects the fully-unrolled instantiation.
    auto gemm = [&](const half_t* A, const half_t* W, const float* bias, const float* rs,
                    float* oF, half_t* oH, int M, int N, int K, int act) {
        dim3 grid(M / 64), blk(128);
        switch (N / 16) {
        case 4:  k_gemm<4> <<<grid, blk, 0, stream>>>(A, W, bias, rs, oF, oH, M, K, act); break;
        case 8:  k_gemm<8> <<<grid, blk, 0, stream>>>(A, W, bias, rs, oF, oH, M, K, act); break;
        default: k_gemm<12><<<grid, blk, 0, stream>>>(A, W, bias, rs, oF, oH, M, K, act); break;
        }
    };
    const int EB = (E_EDGES + 255) / 256;
    const int EH = (E_EDGES * H + 255) / 256;
    const int NH = (NN * H + 255) / 256;

    // ---------------- edge geometry + RBF ----------------
    k_edge_geom<<<EB, 256, 0, stream>>>(positions, cells, ei, eo, batch, bC, bVn, bEattr);
    k_gather_z<<<(E_EDGES * 128 + 255) / 256, 256, 0, stream>>>(ei, numbers, P(6), bZcat);

    // ---------------- tensor embedding ----------------
    gemm(bZcat,  w_emb2, P(8),  nullptr, bZij, nullptr, E_EDGES, 64, 128, 0);
    gemm(bEattr, w_dp1,  P(10), bC,      bW1,  nullptr, E_EDGES, 64, 32, 0);
    gemm(bEattr, w_dp2,  P(12), bC,      bW2,  nullptr, E_EDGES, 64, 32, 0);
    gemm(bEattr, w_dp3,  P(14), bC,      bW3,  nullptr, E_EDGES, 64, 32, 0);
    hipMemsetAsync(bcI, 0, (size_t)NN * 64 * 4, stream);
    hipMemsetAsync(bcA, 0, (size_t)NN * 192 * 4, stream);
    hipMemsetAsync(bcS, 0, (size_t)NN * 384 * 4, stream);
    k_embed_scatter<<<EH, 256, 0, stream>>>(ei, bZij, bW1, bW2, bW3, bVn, bcI, bcA, bcS);
    k_node_norm_embed<<<NH, 256, 0, stream>>>(bcI, bcA, bcS, bNorm, bcI16, bcA16, bcS16);
    k_layernorm<<<(NN + 255) / 256, 256, 0, stream>>>(bNorm, P(15), P(16), bLn16, NN, 64);
    gemm(bLn16,  w_s1, P(18), nullptr, nullptr, bM1_16, NN, 128, 64, 1);
    gemm(bM1_16, w_s2, P(20), nullptr, bM2f, nullptr, NN, 192, 128, 1);
    gemm(bcI16, w_embt,        nullptr, nullptr, bdI, nullptr, NN,     64, 64, 0);
    gemm(bcA16, w_embt + 4096, nullptr, nullptr, bdA, nullptr, NN * 3, 64, 64, 0);
    gemm(bcS16, w_embt + 8192, nullptr, nullptr, bdS, nullptr, NN * 6, 64, 64, 0);
    k_build_X<<<NH, 256, 0, stream>>>(bdI, bdA, bdS, bM2f, bX);

    // ---------------- TensorConv layers ----------------
    for (int l = 0; l < 2; ++l) {
        int b = 22 + l * 7;
        gemm(bEattr, w_cs1[l], P(b + 1), nullptr, nullptr, bEa1, E_EDGES, 64, 32, 1);
        gemm(bEa1,   w_cs2[l], P(b + 3), nullptr, nullptr, bEa2, E_EDGES, 128, 64, 1);
        gemm(bEa2,   w_cs3[l], P(b + 5), bC,      bEaF, nullptr, E_EDGES, 192, 128, 1);
        k_norm_decomp<<<NH, 256, 0, stream>>>(bX, bcI16, bcA16, bcS16);
        gemm(bcI16, w_ct[l],        nullptr, nullptr, bdI, nullptr, NN,     64, 64, 0);
        gemm(bcA16, w_ct[l] + 4096, nullptr, nullptr, bdA, nullptr, NN * 3, 64, 64, 0);
        gemm(bcS16, w_ct[l] + 8192, nullptr, nullptr, bdS, nullptr, NN * 6, 64, 64, 0);
        k_build_Y<<<NH, 256, 0, stream>>>(bdI, bdA, bdS, bY);
        hipMemsetAsync(bMsg, 0, (size_t)NN * 576 * 4, stream);
        k_msg_scatter<<<EH, 256, 0, stream>>>(ei, bEaF, bdI, bdA, bdS, bMsg);
        k_conv_update1<<<NH, 256, 0, stream>>>(bMsg, bY, bcI16, bcA16, bcS16);
        gemm(bcI16, w_ct[l] + 12288, nullptr, nullptr, bdI, nullptr, NN,     64, 64, 0);
        gemm(bcA16, w_ct[l] + 16384, nullptr, nullptr, bdA, nullptr, NN * 3, 64, 64, 0);
        gemm(bcS16, w_ct[l] + 20480, nullptr, nullptr, bdS, nullptr, NN * 6, 64, 64, 0);
        k_conv_update2<<<NH, 256, 0, stream>>>(bdI, bdA, bdS, bX);
    }

    // ---------------- readout ----------------
    k_readout_feat<<<NH, 256, 0, stream>>>(bX, bFeat);
    k_layernorm<<<(NN + 255) / 256, 256, 0, stream>>>(bFeat, P(36), P(37), bFeat16, NN, 192);
    gemm(bFeat16, w_mlp1, P(39), nullptr, nullptr, bH1, NN, 64, 192, 1);
    gemm(bH1,     w_mlp2, P(41), nullptr, bH2, nullptr, NN, 64, 64, 1);
    hipMemsetAsync(d_out, 0, (size_t)NGRAPH * 4, stream);
    k_final<<<(NN + 255) / 256, 256, 0, stream>>>(bH2, P(42), P(43), batch, (float*)d_out);
#undef P
}
